// WindowMultiheadPosAttention_17480516894743
// MI455X (gfx1250) — matrix operand order repaired and checked by reading, not compiled, not run on hardware
//
#include <hip/hip_runtime.h>
#include <hip/hip_bf16.h>

typedef _Float16 v16h __attribute__((ext_vector_type(16)));
typedef float    v8f  __attribute__((ext_vector_type(8)));

union HFrag { v16h v; unsigned u[8]; _Float16 h[16]; };

constexpr int E      = 768;
constexpr int NTOK   = 4096;
constexpr int BATCH  = 2;
constexpr int MROWS  = BATCH * NTOK;   // 8192
constexpr int GROUPS = 48;             // heads * n_win
constexpr int LSEQ   = 1024;           // tokens per window group
constexpr int HD     = 64;
constexpr float LOG2E = 1.44269504088896340f;

// ---------------------------------------------------------------- converts
__global__ void cvt_f32_f16(const float* __restrict__ in,
                            _Float16* __restrict__ out, int n) {
  int i = blockIdx.x * 256 + threadIdx.x;
  if (i < n) out[i] = (_Float16)in[i];
}

// W[k][n] (f32, row-major) -> Wt[n][k] (f16, K contiguous). Coalesced writes.
__global__ void cvt_transpose_f16(const float* __restrict__ in,
                                  _Float16* __restrict__ out) {
  const int k = blockIdx.x * 32 + (threadIdx.x & 31);
  const int n = blockIdx.y * 8 + (threadIdx.x >> 5);
  out[(size_t)n * E + k] = (_Float16)in[(size_t)k * E + n];
}

// ------------------------------------------------------- async-copy helpers
__device__ __forceinline__ void async_b128(unsigned lds_addr, const void* gptr) {
  asm volatile("global_load_async_to_lds_b128 %0, %1, off"
               :: "v"(lds_addr), "v"(gptr) : "memory");
}
__device__ __forceinline__ void wait_async0() {
  asm volatile("s_wait_asynccnt 0" ::: "memory");
}
__device__ __forceinline__ unsigned lds_off(const void* p) {
  return (unsigned)(unsigned long long)p;   // LDS aperture: offset in addr[31:0]
}

// ---------------------------------------------------------------- GEMM
// C[8192 x 768] = X[8192 x 768] @ W + bias. X f16 row-major, Wt f16 [N][K].
// Block: 256 thr (8 waves), tile 128x64; wave tile 32x32 (4 WMMA / K-step).
// Double-buffered LDS staged with async global->LDS copies (ASYNCcnt).
// mode 0: q -> outH separated [grp][p][d], *0.625   mode 1: k -> same
// mode 2: v -> outH transposed-separated [grp][d][p] mode 3: -> outF f32
__global__ __launch_bounds__(256)
void gemm_wmma(const _Float16* __restrict__ X,
               const _Float16* __restrict__ Wt,
               const float* __restrict__ bias,
               _Float16* __restrict__ outH,
               float* __restrict__ outF,
               int mode) {
  __shared__ __align__(16) unsigned As[2][128 * 16];  // 128 rows x 32 halves
  __shared__ __align__(16) unsigned Bs[2][64 * 16];   // 64 n-rows x 32 halves

  const int tid  = threadIdx.x;
  const int lane = tid & 31;
  const int wave = tid >> 5;
  const int half = lane >> 4;
  const int l16  = lane & 15;
  const int row0 = blockIdx.y * 128;
  const int n0   = blockIdx.x * 64;
  const int wm   = wave & 3;   // 4 row tiles of 32
  const int wn   = wave >> 2;  // 2 col tiles of 32

  // per-thread staging coordinates
  const int ar = tid >> 1;             // A: row 0..127
  const int ac = (tid & 1) << 3;       // A: uint col 0 or 8 (two b128 each)
  const int bn = tid >> 2;             // B: n-row 0..63
  const int bc = (tid & 3) << 2;       // B: uint col 0,4,8,12 (one b128)

  auto stage = [&](int buf, int kc) {
    const _Float16* ga = X + (size_t)(row0 + ar) * E + kc + ac * 2;
    async_b128(lds_off(&As[buf][ar * 16 + ac]),     ga);
    async_b128(lds_off(&As[buf][ar * 16 + ac + 4]), ga + 8);
    const _Float16* gb = Wt + (size_t)(n0 + bn) * E + kc + bc * 2;
    async_b128(lds_off(&Bs[buf][bn * 16 + bc]),     gb);
  };

  v8f acc[2][2] = {};

  stage(0, 0);
  wait_async0();
  __syncthreads();

  for (int kc = 0; kc < E; kc += 32) {
    const int cur = (kc >> 5) & 1;
    const bool more = (kc + 32) < E;
    if (more) stage(cur ^ 1, kc + 32);

    HFrag a[2], b[2];
#pragma unroll
    for (int t = 0; t < 2; ++t) {
      const int arow = wm * 32 + t * 16 + l16;
      const int brow = wn * 32 + t * 16 + l16;
#pragma unroll
      for (int v = 0; v < 8; ++v) {
        const int ka = 16 * half + 2 * v;                       // A swizzle
        a[t].u[v] = As[cur][arow * 16 + (ka >> 1)];
        const int kb = 16 * half + 2 * v;                       // B swizzle
        b[t].u[v] = Bs[cur][brow * 16 + (kb >> 1)];
      }
    }
#pragma unroll
    for (int i = 0; i < 2; ++i)
#pragma unroll
      for (int j = 0; j < 2; ++j)
        acc[i][j] = __builtin_amdgcn_wmma_f32_16x16x32_f16(
            false, a[i].v, false, b[j].v, (short)0, acc[i][j], false, false);

    if (more) wait_async0();
    __syncthreads();
  }

  // --- epilogue ---
#pragma unroll
  for (int i = 0; i < 2; ++i) {
#pragma unroll
    for (int j = 0; j < 2; ++j) {
      const int N = n0 + wn * 32 + j * 16 + l16;
      const float bvv = bias[N];
      const int mbase = row0 + wm * 32 + i * 16 + 8 * half;
#pragma unroll
      for (int r = 0; r < 8; ++r) {
        const int gr = mbase + r;
        float val = acc[i][j][r] + bvv;
        if (mode == 3) {
          outF[(size_t)gr * E + N] = val;
        } else {
          if (mode == 0) val *= 0.625f;   // SCALE * SOFTMAX_SCALE
          const int bI   = gr >> 12;      // / 4096
          const int tok  = gr & 4095;
          const int head = N >> 6;
          const int dd   = N & 63;
          const int rr = tok >> 6, cc = tok & 63;
          const int g = head * 4 + (rr >> 5) * 2 + (cc >> 5);
          const int p = (rr & 31) * 32 + (cc & 31);
          const size_t grp = (size_t)bI * GROUPS + g;
          if (mode == 2)
            outH[(grp * HD + dd) * LSEQ + p] = (_Float16)val;   // V transposed
          else
            outH[(grp * LSEQ + p) * HD + dd] = (_Float16)val;   // Q / K
        }
      }
    }
  }
}

// ---------------------------------------------------------------- attention
// grid (16, 96): x = 64-row Q chunk, y = group (b*48+g). 4 waves/block,
// each wave: 16 Q rows x full d=64, flash-style over 1024 keys in 32-chunks.
__global__ __launch_bounds__(128)
void attn_wmma(const _Float16* __restrict__ Q,
               const _Float16* __restrict__ K,
               const _Float16* __restrict__ Vt,
               _Float16* __restrict__ AO) {
  __shared__ _Float16 Ps[4][16][32];   // per-wave P staging (C->A relayout)

  const int tid  = threadIdx.x;
  const int lane = tid & 31;
  const int wave = tid >> 5;
  const int half = lane >> 4;
  const int l16  = lane & 15;
  const int grp  = blockIdx.y;
  const int q0   = blockIdx.x * 64 + wave * 16;

  const _Float16* Qg = Q  + (size_t)grp * LSEQ * HD;
  const _Float16* Kg = K  + (size_t)grp * LSEQ * HD;
  const _Float16* Vg = Vt + (size_t)grp * HD * LSEQ;

  // Q A-fragments: 2 K-slices of d=64, kept in registers for whole kernel
  HFrag aq[2];
#pragma unroll
  for (int s = 0; s < 2; ++s)
#pragma unroll
    for (int v = 0; v < 8; ++v) {
      const int ka = 16 * half + 2 * v;
      aq[s].u[v] = *(const unsigned*)(Qg + (size_t)(q0 + l16) * HD + s * 32 + ka);
    }

  v8f o[4] = {};
  float mrun[8], lrun[8];
#pragma unroll
  for (int i = 0; i < 8; ++i) { mrun[i] = -3.0e38f; lrun[i] = 0.f; }

  for (int kb = 0; kb < LSEQ; kb += 32) {
    // K B-fragments: 2 col tiles x 2 K-slices (K rows are d-contiguous)
    HFrag bk[2][2];
#pragma unroll
    for (int ntile = 0; ntile < 2; ++ntile)
#pragma unroll
      for (int s = 0; s < 2; ++s)
#pragma unroll
        for (int v = 0; v < 8; ++v) {
          const int kk = 16 * half + 2 * v;
          bk[ntile][s].u[v] = *(const unsigned*)
              (Kg + (size_t)(kb + ntile * 16 + l16) * HD + s * 32 + kk);
        }
    v8f s0 = {}, s1 = {};
    s0 = __builtin_amdgcn_wmma_f32_16x16x32_f16(false, aq[0].v, false, bk[0][0].v, (short)0, s0, false, false);
    s0 = __builtin_amdgcn_wmma_f32_16x16x32_f16(false, aq[1].v, false, bk[0][1].v, (short)0, s0, false, false);
    s1 = __builtin_amdgcn_wmma_f32_16x16x32_f16(false, aq[0].v, false, bk[1][0].v, (short)0, s1, false, false);
    s1 = __builtin_amdgcn_wmma_f32_16x16x32_f16(false, aq[1].v, false, bk[1][1].v, (short)0, s1, false, false);

    // online softmax per row (rows stripe: VGPR i, lanes split by half)
#pragma unroll
    for (int i = 0; i < 8; ++i) {
      float a = s0[i], b = s1[i];
      float mx = fmaxf(a, b);
#pragma unroll
      for (int msk = 1; msk < 16; msk <<= 1) mx = fmaxf(mx, __shfl_xor(mx, msk, 32));
      const float mn = fmaxf(mrun[i], mx);
      const float alpha = exp2f((mrun[i] - mn) * LOG2E);
      const float p0 = exp2f((a - mn) * LOG2E);
      const float p1 = exp2f((b - mn) * LOG2E);
      float rs = p0 + p1;
#pragma unroll
      for (int msk = 1; msk < 16; msk <<= 1) rs += __shfl_xor(rs, msk, 32);
      lrun[i] = lrun[i] * alpha + rs;
      mrun[i] = mn;
#pragma unroll
      for (int t = 0; t < 4; ++t) o[t][i] *= alpha;
      const int mrow = i + 8 * half;
      Ps[wave][mrow][l16]      = (_Float16)p0;
      Ps[wave][mrow][16 + l16] = (_Float16)p1;
    }
    asm volatile("s_wait_dscnt 0" ::: "memory");  // LDS RAW before relayout read

    // P C-layout -> A-layout fragment
    HFrag ap;
#pragma unroll
    for (int v = 0; v < 8; ++v) {
      const int ka = 16 * half + 2 * v;
      ap.h[2 * v]     = Ps[wave][l16][ka];
      ap.h[2 * v + 1] = Ps[wave][l16][ka + 1];
    }

    // O += P @ V   (Vt rows are p-contiguous => perfect B layout)
#pragma unroll
    for (int t = 0; t < 4; ++t) {
      HFrag bv;
#pragma unroll
      for (int v = 0; v < 8; ++v) {
        const int kk = 16 * half + 2 * v;
        bv.u[v] = *(const unsigned*)
            (Vg + (size_t)(t * 16 + l16) * LSEQ + kb + kk);
      }
      o[t] = __builtin_amdgcn_wmma_f32_16x16x32_f16(
          false, ap.v, false, bv.v, (short)0, o[t], false, false);
    }
  }

  // epilogue: reference merges windows as a RAW view: token' = win*1024 + p
  const int bI   = grp / GROUPS;
  const int g    = grp % GROUPS;
  const int head = g >> 2;
  const int win  = g & 3;
#pragma unroll
  for (int i = 0; i < 8; ++i) {
    const int p   = q0 + i + 8 * half;
    const int tok = win * LSEQ + p;
    const float inv = 1.0f / lrun[i];
#pragma unroll
    for (int t = 0; t < 4; ++t) {
      const int e = head * HD + t * 16 + l16;
      AO[((size_t)bI * NTOK + tok) * E + e] = (_Float16)(o[t][i] * inv);
    }
  }
}

// ---------------------------------------------------------------- launch
extern "C" void kernel_launch(void* const* d_in, const int* in_sizes, int n_in,
                              void* d_out, int out_size, void* d_ws, size_t ws_size,
                              hipStream_t stream) {
  const float* x   = (const float*)d_in[0];
  const float* pe  = (const float*)d_in[1];
  const float* Wq  = (const float*)d_in[2];
  const float* bq  = (const float*)d_in[3];
  const float* Wk  = (const float*)d_in[4];
  const float* bk  = (const float*)d_in[5];
  const float* Wv  = (const float*)d_in[6];
  const float* bv  = (const float*)d_in[7];
  const float* Wp  = (const float*)d_in[8];
  const float* bp  = (const float*)d_in[9];

  constexpr size_t ACT = (size_t)MROWS * E;     // 6291456 elems
  constexpr size_t WEL = (size_t)E * E;         // 589824 elems
  constexpr size_t SEP = (size_t)BATCH * GROUPS * LSEQ * HD;  // == ACT

  char* ws = (char*)d_ws;
  size_t off = 0;
  _Float16* xh  = (_Float16*)(ws + off); off += ACT * 2;
  _Float16* ph  = (_Float16*)(ws + off); off += ACT * 2;
  _Float16* wqh = (_Float16*)(ws + off); off += WEL * 2;   // transposed [N][K]
  _Float16* wkh = (_Float16*)(ws + off); off += WEL * 2;
  _Float16* wvh = (_Float16*)(ws + off); off += WEL * 2;
  _Float16* wph = (_Float16*)(ws + off); off += WEL * 2;
  _Float16* qs  = (_Float16*)(ws + off); off += SEP * 2;
  _Float16* ks  = (_Float16*)(ws + off); off += SEP * 2;
  _Float16* vt  = (_Float16*)(ws + off); off += SEP * 2;
  _Float16* ao  = (_Float16*)(ws + off); off += SEP * 2;
  if (ws_size < off) return;

  cvt_f32_f16<<<(int)((ACT + 255) / 256), 256, 0, stream>>>(x,  xh, (int)ACT);
  cvt_f32_f16<<<(int)((ACT + 255) / 256), 256, 0, stream>>>(pe, ph, (int)ACT);
  dim3 tgrid(E / 32, E / 8);   // (24, 96)
  cvt_transpose_f16<<<tgrid, 256, 0, stream>>>(Wq, wqh);
  cvt_transpose_f16<<<tgrid, 256, 0, stream>>>(Wk, wkh);
  cvt_transpose_f16<<<tgrid, 256, 0, stream>>>(Wv, wvh);
  cvt_transpose_f16<<<tgrid, 256, 0, stream>>>(Wp, wph);

  dim3 ggrid(E / 64, MROWS / 128);  // (12, 64)
  gemm_wmma<<<ggrid, 256, 0, stream>>>(ph, wqh, bq, qs, nullptr, 0);
  gemm_wmma<<<ggrid, 256, 0, stream>>>(xh, wkh, bk, ks, nullptr, 1);
  gemm_wmma<<<ggrid, 256, 0, stream>>>(xh, wvh, bv, vt, nullptr, 2);

  attn_wmma<<<dim3(LSEQ / 64, BATCH * GROUPS), 128, 0, stream>>>(qs, ks, vt, ao);

  gemm_wmma<<<ggrid, 256, 0, stream>>>(ao, wph, bp, nullptr, (float*)d_out, 3);
}
